// ConditioningEncoder_81286551044411
// MI455X (gfx1250) — matrix operand order repaired and checked by reading, not compiled, not run on hardware
//
#include <hip/hip_runtime.h>
#include <math.h>

// Problem constants (from reference)
#define BSZ     4
#define INC     100
#define E_DIM   1024
#define L_DIM   1024
#define H_DIM   16
#define NBLK    4
#define NGROUPS 32
#define D_HEAD  64
#define QKV     (3 * E_DIM)     // 3072
#define EPS_GN  1e-5f
#define MASK_VAL (-3.402823466e+38f)

typedef __attribute__((ext_vector_type(16))) __bf16 v16bf;
typedef __attribute__((ext_vector_type(8)))  float  v8f;
typedef int v4i_vs __attribute__((vector_size(16)));   // matches builtin's pointee type

// ---------- small helpers ----------
__device__ __forceinline__ unsigned short f2bf(float f) {
    unsigned int u = __float_as_uint(f);
    u += 0x7FFFu + ((u >> 16) & 1u);     // round-to-nearest-even
    return (unsigned short)(u >> 16);
}
__device__ __forceinline__ float bf2f(unsigned short b) {
    return __uint_as_float(((unsigned int)b) << 16);
}
__device__ __forceinline__ v8f zero8() {
    v8f z = {0.f, 0.f, 0.f, 0.f, 0.f, 0.f, 0.f, 0.f};
    return z;
}
__device__ __forceinline__ v8f wmma_bf16(v16bf a, v16bf b, v8f c) {
    return __builtin_amdgcn_wmma_f32_16x16x32_bf16(
        /*neg_a=*/false, a, /*neg_b=*/false, b,
        /*c_mod=*/(short)0, c, /*reuse_a=*/false, /*reuse_b=*/false);
}

// Async global->LDS staging (CDNA5 GLOBAL_LOAD_ASYNC_TO_LDS_B128, ASYNCcnt-tracked)
// Builtin signature (from toolchain diagnostic): (v4i AS1* gsrc, v4i AS3* ldst, int off, int cpol)
__device__ __forceinline__ void async_g2l_b128(const unsigned short* g, unsigned short* l) {
#if __has_builtin(__builtin_amdgcn_global_load_async_to_lds_b128)
    __builtin_amdgcn_global_load_async_to_lds_b128(
        (__attribute__((address_space(1))) v4i_vs*)(unsigned long long)g,
        (__attribute__((address_space(3))) v4i_vs*)l, 0, 0);
#else
    unsigned loff = (unsigned)(unsigned long long)
        (__attribute__((address_space(3))) unsigned short*)l;
    unsigned long long ga = (unsigned long long)g;
    asm volatile("global_load_async_to_lds_b128 %0, %1, off"
                 :: "v"(loff), "v"(ga) : "memory");
#endif
}
__device__ __forceinline__ void wait_async_le1() {
#if __has_builtin(__builtin_amdgcn_s_wait_asynccnt)
    __builtin_amdgcn_s_wait_asynccnt(1);
#else
    asm volatile("s_wait_asynccnt 0x1" ::: "memory");
#endif
}
__device__ __forceinline__ void wait_async_le0() {
#if __has_builtin(__builtin_amdgcn_s_wait_asynccnt)
    __builtin_amdgcn_s_wait_asynccnt(0);
#else
    asm volatile("s_wait_asynccnt 0x0" ::: "memory");
#endif
}

// A-fragment (16x32, bf16, M x K): lane<16 holds row m=lane, K = k..k+15;
// lane>=16 holds row m=lane-16, K = k+16..k+31.
__device__ __forceinline__ v16bf load_a_frag(const unsigned short* base, size_t stride,
                                             int row, int k) {
    int lane = threadIdx.x & 31;
    int lh = lane >> 4, ln = lane & 15;
    const unsigned short* p = base + (size_t)(row + ln) * stride + k;
    v16bf f;
    __builtin_memcpy(&f, p + lh * 16, 32);
    return f;
}
// B-fragment (32x16, bf16, K x N) loaded from row-major R (N x K), R[n][k] = B[k][n]:
// lane<16 holds column n=lane with K=0..15; lane>=16 holds column n=lane-16 with K=16..31.
__device__ __forceinline__ v16bf load_b_frag(const unsigned short* base, size_t stride,
                                             int nrow, int k) {
    int lane = threadIdx.x & 31;
    int lh = lane >> 4, ln = lane & 15;
    const unsigned short* p = base + (size_t)(nrow + ln) * stride + k + lh * 16;
    v16bf f;
    __builtin_memcpy(&f, p, 32);
    return f;
}
// Same B-fragment pattern but from an LDS tile of 64 rows x 32 k (row stride 32)
__device__ __forceinline__ v16bf load_b_frag_lds(const unsigned short* buf, int nrow) {
    int lane = threadIdx.x & 31;
    int lh = lane >> 4, ln = lane & 15;
    const unsigned short* p = buf + (nrow + ln) * 32 + lh * 16;
    v16bf f;
    __builtin_memcpy(&f, p, 32);
    return f;
}
// reductions confined to 16-lane halves of the wave32
__device__ __forceinline__ float halfmax(float v) {
    v = fmaxf(v, __shfl_xor(v, 1));
    v = fmaxf(v, __shfl_xor(v, 2));
    v = fmaxf(v, __shfl_xor(v, 4));
    v = fmaxf(v, __shfl_xor(v, 8));
    return v;
}
__device__ __forceinline__ float halfsum(float v) {
    v += __shfl_xor(v, 1);
    v += __shfl_xor(v, 2);
    v += __shfl_xor(v, 4);
    v += __shfl_xor(v, 8);
    return v;
}

// ---------- kernel 1: fp32 weights -> bf16 scratch copies ----------
__global__ void convert_weights_kernel(const float* qkvw, const float* outw,
                                       unsigned short* qbf, unsigned short* obf) {
    const size_t NQ = (size_t)NBLK * QKV * E_DIM;
    const size_t NO = (size_t)NBLK * E_DIM * E_DIM;
    size_t idx = (size_t)blockIdx.x * blockDim.x + threadIdx.x;
    if (idx < NQ)            qbf[idx]      = f2bf(qkvw[idx]);
    else if (idx < NQ + NO)  obf[idx - NQ] = f2bf(outw[idx - NQ]);
}

// ---------- kernel 2: h = init_w @ x + init_b  (K=100, tiny; fp32 direct) ----------
__global__ void init_gemm_kernel(const float* __restrict__ x, const float* __restrict__ w,
                                 const float* __restrict__ bias, float* __restrict__ h) {
    size_t idx = (size_t)blockIdx.x * blockDim.x + threadIdx.x;
    if (idx >= (size_t)BSZ * E_DIM * L_DIM) return;
    int l = (int)(idx % L_DIM);
    size_t t = idx / L_DIM;
    int o = (int)(t % E_DIM);
    int b = (int)(t / E_DIM);
    float acc = bias[o];
    const float* xr = x + (size_t)b * INC * L_DIM + l;
    const float* wr = w + (size_t)o * INC;
    #pragma unroll 4
    for (int i = 0; i < INC; ++i) acc += wr[i] * xr[(size_t)i * L_DIM];
    h[idx] = acc;
}

// ---------- kernel 3: group norm; writes hn (f32, B,E,L) and hnT (bf16, B,L,E) ----------
__global__ void groupnorm_kernel(const float* __restrict__ h,
                                 const float* __restrict__ nw, const float* __restrict__ nb,
                                 float* __restrict__ hn, unsigned short* __restrict__ hnT) {
    const int CG = E_DIM / NGROUPS;          // 32 channels per group
    const int N = CG * L_DIM;                // 32768 elems per group
    int b = blockIdx.x / NGROUPS, g = blockIdx.x % NGROUPS;
    const float* src = h + ((size_t)b * E_DIM + g * CG) * L_DIM;
    __shared__ float s1[256], s2[256];
    float a1 = 0.f, a2 = 0.f;
    for (int i = threadIdx.x; i < N; i += blockDim.x) {
        float v = src[i]; a1 += v; a2 += v * v;
    }
    s1[threadIdx.x] = a1; s2[threadIdx.x] = a2;
    __syncthreads();
    for (int off = 128; off > 0; off >>= 1) {
        if ((int)threadIdx.x < off) {
            s1[threadIdx.x] += s1[threadIdx.x + off];
            s2[threadIdx.x] += s2[threadIdx.x + off];
        }
        __syncthreads();
    }
    float mu  = s1[0] / (float)N;
    float var = s2[0] / (float)N - mu * mu;
    float inv = rsqrtf(var + EPS_GN);
    for (int i = threadIdx.x; i < N; i += blockDim.x) {
        int cl = i / L_DIM, l = i % L_DIM;
        int c = g * CG + cl;
        float y = (src[i] - mu) * inv * nw[c] + nb[c];
        hn[((size_t)b * E_DIM + c) * L_DIM + l] = y;
        hnT[((size_t)b * L_DIM + l) * E_DIM + c] = f2bf(y);
    }
}

// ---------- kernel 4: WMMA GEMM  D[l,o] = sum_c AT[l,c] * W[o,c] (+bias) ----------
// Per block: 8 waves, all sharing one 64-col weight group. The 64x32 bf16 W tile is
// staged to LDS with GLOBAL_LOAD_ASYNC_TO_LDS_B128 (double-buffered, ASYNCcnt-synced);
// each wave then computes a 32(M)x64(N) tile: 8 WMMA per k-step, A from global,
// B broadcast from LDS.
// mode 0: write bf16 outT (stride O)          [QKV projection]
// mode 1: write f32 dst = resid + D (B,E,L)   [output projection + residual]
__global__ __launch_bounds__(256)
void gemm_bt_kernel(const unsigned short* __restrict__ AT, const unsigned short* __restrict__ W,
                    const float* __restrict__ bias, int O, int mode,
                    unsigned short* __restrict__ outT, float* __restrict__ dstF,
                    const float* __restrict__ resid) {
    int lane = threadIdx.x & 31;
    int lh = lane >> 4, ln = lane & 15;
    int wave = blockIdx.x * 8 + (threadIdx.x >> 5);
    int mt = wave & 31;                       // L/32 = 32 row tiles of 32
    int ng = (blockIdx.x * 8) >> 5;           // weight-col group: uniform across block
    int b  = blockIdx.y;

    __shared__ __align__(16) unsigned short wtile[2][64 * 32];   // 2 x 4KB double buffer

    // one async b128 per thread stages a full 64x32 bf16 tile (256 * 16B = 4KB)
    int sr = threadIdx.x >> 2;                // row 0..63
    int sc = (threadIdx.x & 3) * 8;           // col {0,8,16,24}
    const unsigned short* wrow = W + (size_t)(ng * 64 + sr) * E_DIM + sc;

    const unsigned short* ATb = AT + (size_t)b * L_DIM * E_DIM;
    v8f acc[2][4] = {{zero8(), zero8(), zero8(), zero8()},
                     {zero8(), zero8(), zero8(), zero8()}};

    async_g2l_b128(wrow, &wtile[0][sr * 32 + sc]);     // stage k=0

    for (int k = 0; k < E_DIM; k += 32) {
        int cur = (k >> 5) & 1;
        bool has_next = (k + 32 < E_DIM);
        if (has_next) {
            async_g2l_b128(wrow + k + 32, &wtile[cur ^ 1][sr * 32 + sc]);
            wait_async_le1();                 // current tile's async op complete
        } else {
            wait_async_le0();
        }
        __syncthreads();                      // current tile visible to all waves

        v16bf a0 = load_a_frag(ATb, E_DIM, mt * 32, k);
        v16bf a1 = load_a_frag(ATb, E_DIM, mt * 32 + 16, k);
        if (k + 128 < E_DIM) {                // keep the per-wave A stream ahead in cache
            __builtin_prefetch(ATb + (size_t)(mt * 32 + ln) * E_DIM + k + 128, 0, 1);
        }
        #pragma unroll
        for (int j = 0; j < 4; ++j) {
            v16bf bf = load_b_frag_lds(&wtile[cur][0], j * 16);
            acc[0][j] = wmma_bf16(a0, bf, acc[0][j]);
            acc[1][j] = wmma_bf16(a1, bf, acc[1][j]);
        }
        __syncthreads();                      // all reads done before buffer is re-staged
    }

    #pragma unroll
    for (int j = 0; j < 4; ++j) {
        int o = ng * 64 + j * 16 + ln;
        float bb = bias[o];
        #pragma unroll
        for (int mi = 0; mi < 2; ++mi) {
            #pragma unroll
            for (int r = 0; r < 8; ++r) {
                int row = mt * 32 + mi * 16 + r + 8 * lh;
                float v = acc[mi][j][r] + bb;
                if (mode == 0) {
                    outT[((size_t)b * L_DIM + row) * O + o] = f2bf(v);
                } else {
                    size_t di = ((size_t)b * E_DIM + o) * L_DIM + row;
                    dstF[di] = resid[di] + v;
                }
            }
        }
    }
}

// ---------- kernel 5: vT[b,h,dd,l] = V slice of qkvT (for contiguous PV B-frags) ----------
__global__ void transpose_v_kernel(const unsigned short* __restrict__ qkvT,
                                   unsigned short* __restrict__ vT) {
    size_t idx = (size_t)blockIdx.x * blockDim.x + threadIdx.x;
    if (idx >= (size_t)BSZ * H_DIM * D_HEAD * L_DIM) return;
    int l = (int)(idx % L_DIM);
    size_t t = idx / L_DIM;
    int dd = (int)(t % D_HEAD); t /= D_HEAD;
    int hh = (int)(t % H_DIM);
    int b  = (int)(t / H_DIM);
    vT[idx] = qkvT[((size_t)b * L_DIM + l) * QKV + hh * 192 + 128 + dd];
}

// ---------- kernel 6: flash attention, one wave per (b, head, 16 queries) ----------
__global__ __launch_bounds__(32)
void attention_kernel(const unsigned short* __restrict__ qkvT,
                      const unsigned short* __restrict__ vT,
                      const unsigned char* __restrict__ mask,
                      unsigned short* __restrict__ ctxT) {
    int q0 = blockIdx.x * 16;
    int hh = blockIdx.y;
    int b  = blockIdx.z;
    int lane = threadIdx.x & 31;
    int lh = lane >> 4, ln = lane & 15;

    const unsigned short* qbase = qkvT + (size_t)b * L_DIM * QKV + hh * 192;       // q rows
    const unsigned short* kbase = qbase + 64;                                      // k rows
    const unsigned short* vtb   = vT + (size_t)(b * H_DIM + hh) * D_HEAD * L_DIM;  // (d,L)

    v16bf aq0 = load_a_frag(qbase, QKV, q0, 0);
    v16bf aq1 = load_a_frag(qbase, QKV, q0, 32);

    v8f O0 = zero8(), O1 = zero8(), O2 = zero8(), O3 = zero8();
    float rm[8], rs[8];
    #pragma unroll
    for (int r = 0; r < 8; ++r) { rm[r] = -INFINITY; rs[r] = 0.f; }

    __shared__ __align__(32) unsigned short pl[16 * 32];
    const float sc2 = 1.0f / (float)D_HEAD;   // (1/sqrt(d))^2

    for (int kc = 0; kc < L_DIM / 32; ++kc) {
        int kb = kc * 32;
        v8f s0 = zero8(), s1 = zero8();
        s0 = wmma_bf16(aq0, load_b_frag(kbase, QKV, kb,       0),  s0);
        s0 = wmma_bf16(aq1, load_b_frag(kbase, QKV, kb,       32), s0);
        s1 = wmma_bf16(aq0, load_b_frag(kbase, QKV, kb + 16,  0),  s1);
        s1 = wmma_bf16(aq1, load_b_frag(kbase, QKV, kb + 16,  32), s1);

        bool mk0 = mask[(size_t)b * L_DIM + kb + ln] != 0;
        bool mk1 = mask[(size_t)b * L_DIM + kb + 16 + ln] != 0;

        #pragma unroll
        for (int r = 0; r < 8; ++r) {
            float v0 = mk0 ? MASK_VAL : s0[r] * sc2;
            float v1 = mk1 ? MASK_VAL : s1[r] * sc2;
            float tm = halfmax(fmaxf(v0, v1));
            float nm = fmaxf(rm[r], tm);
            float alpha = __expf(rm[r] - nm);
            float p0 = __expf(v0 - nm);
            float p1 = __expf(v1 - nm);
            rs[r] = rs[r] * alpha + halfsum(p0 + p1);
            rm[r] = nm;
            O0[r] *= alpha; O1[r] *= alpha; O2[r] *= alpha; O3[r] *= alpha;
            int m = r + 8 * lh;
            pl[m * 32 + ln]      = f2bf(p0);
            pl[m * 32 + 16 + ln] = f2bf(p1);
        }
        __syncthreads();
        v16bf pa;
        {
            const unsigned short* pp = pl + ln * 32;     // A row = lane&15
            __builtin_memcpy(&pa, pp + lh * 16, 32);
        }
        O0 = wmma_bf16(pa, load_b_frag(vtb, L_DIM, 0,  kb), O0);
        O1 = wmma_bf16(pa, load_b_frag(vtb, L_DIM, 16, kb), O1);
        O2 = wmma_bf16(pa, load_b_frag(vtb, L_DIM, 32, kb), O2);
        O3 = wmma_bf16(pa, load_b_frag(vtb, L_DIM, 48, kb), O3);
        __syncthreads();
    }

    unsigned short* cb = ctxT + (size_t)b * L_DIM * E_DIM + hh * 64;
    #pragma unroll
    for (int r = 0; r < 8; ++r) {
        float inv = 1.0f / rs[r];
        int row = q0 + r + 8 * lh;
        unsigned short* cr = cb + (size_t)row * E_DIM;
        cr[0  + ln] = f2bf(O0[r] * inv);
        cr[16 + ln] = f2bf(O1[r] * inv);
        cr[32 + ln] = f2bf(O2[r] * inv);
        cr[48 + ln] = f2bf(O3[r] * inv);
    }
}

// ---------- host orchestration ----------
extern "C" void kernel_launch(void* const* d_in, const int* in_sizes, int n_in,
                              void* d_out, int out_size, void* d_ws, size_t ws_size,
                              hipStream_t stream) {
    const float* x      = (const float*)d_in[0];
    const float* init_w = (const float*)d_in[1];
    const float* init_b = (const float*)d_in[2];
    const float* norm_w = (const float*)d_in[3];
    const float* norm_b = (const float*)d_in[4];
    const float* qkv_w  = (const float*)d_in[5];
    const float* qkv_b  = (const float*)d_in[6];
    const float* out_w  = (const float*)d_in[7];
    const float* out_b  = (const float*)d_in[8];
    const unsigned char* mask = (const unsigned char*)d_in[9];
    float* out = (float*)d_out;

    // workspace carve-out (all 256B-aligned slices)
    char* ws = (char*)d_ws;
    size_t off = 0;
    auto carve = [&](size_t bytes) {
        void* p = ws + off;
        off += (bytes + 255) & ~(size_t)255;
        return p;
    };
    float*          h_buf  = (float*)         carve((size_t)BSZ * E_DIM * L_DIM * 4);
    float*          hn_buf = (float*)         carve((size_t)BSZ * E_DIM * L_DIM * 4);
    unsigned short* hnT    = (unsigned short*)carve((size_t)BSZ * L_DIM * E_DIM * 2);
    unsigned short* qkvT   = (unsigned short*)carve((size_t)BSZ * L_DIM * QKV * 2);
    unsigned short* vT     = (unsigned short*)carve((size_t)BSZ * H_DIM * D_HEAD * L_DIM * 2);
    unsigned short* ctxT   = (unsigned short*)carve((size_t)BSZ * L_DIM * E_DIM * 2);
    unsigned short* qkvWbf = (unsigned short*)carve((size_t)NBLK * QKV * E_DIM * 2);
    unsigned short* outWbf = (unsigned short*)carve((size_t)NBLK * E_DIM * E_DIM * 2);
    (void)ws_size; (void)in_sizes; (void)n_in; (void)out_size;

    // weights -> bf16 (once per launch; deterministic)
    {
        size_t total = (size_t)NBLK * QKV * E_DIM + (size_t)NBLK * E_DIM * E_DIM;
        int blocks = (int)((total + 255) / 256);
        convert_weights_kernel<<<blocks, 256, 0, stream>>>(qkv_w, out_w, qkvWbf, outWbf);
    }
    // h = init_w @ x + init_b
    {
        size_t total = (size_t)BSZ * E_DIM * L_DIM;
        int blocks = (int)((total + 255) / 256);
        init_gemm_kernel<<<blocks, 256, 0, stream>>>(x, init_w, init_b, h_buf);
    }

    for (int i = 0; i < NBLK; ++i) {
        // group norm
        groupnorm_kernel<<<BSZ * NGROUPS, 256, 0, stream>>>(
            h_buf, norm_w + (size_t)i * E_DIM, norm_b + (size_t)i * E_DIM, hn_buf, hnT);

        // qkv = qkv_w[i] @ hn + qkv_b[i]   (WMMA + async-LDS weight staging)
        {
            dim3 grid((L_DIM / 32) * (QKV / 64) / 8, BSZ);   // 192 blocks x 8 waves
            gemm_bt_kernel<<<grid, 256, 0, stream>>>(
                hnT, qkvWbf + (size_t)i * QKV * E_DIM, qkv_b + (size_t)i * QKV,
                QKV, 0, qkvT, nullptr, nullptr);
        }
        // V transpose for PV fragments
        {
            size_t total = (size_t)BSZ * H_DIM * D_HEAD * L_DIM;
            transpose_v_kernel<<<(int)((total + 255) / 256), 256, 0, stream>>>(qkvT, vT);
        }
        // flash attention (WMMA QK^T + PV)
        {
            dim3 grid(L_DIM / 16, H_DIM, BSZ);
            attention_kernel<<<grid, 32, 0, stream>>>(qkvT, vT, mask, ctxT);
        }
        // h = hn + out_w[i] @ ctx + out_b[i]  (WMMA + fused residual)
        {
            float* dst = (i == NBLK - 1) ? out : h_buf;
            dim3 grid((L_DIM / 32) * (E_DIM / 64) / 8, BSZ); // 64 blocks x 8 waves
            gemm_bt_kernel<<<grid, 256, 0, stream>>>(
                ctxT, outWbf + (size_t)i * E_DIM * E_DIM, out_b + (size_t)i * E_DIM,
                E_DIM, 1, nullptr, dst, hn_buf);
        }
    }
}
